// GCN_11424613007606
// MI455X (gfx1250) — compile-verified, matrix operand order checked by reading, not run on hardware
//
#include <hip/hip_runtime.h>
#include <hip/hip_bf16.h>

// ---------------------------------------------------------------------------
// GCN (3-layer) for MI455X / gfx1250, wave32 + WMMA bf16.
//   layer: pre = X @ W^T  (WMMA f32_16x16x32_bf16, fp32 accum)
//          out = dinv[dst]*dinv[src] scatter of pre  + dinv[i]^2*pre[i] + bias
//   fp32->bf16 conversion (with fused ReLU) is a separate one-shot pass so
//   the GEMM inner loop is pure load_b128 + v_wmma.
// ---------------------------------------------------------------------------

typedef __attribute__((ext_vector_type(16))) __bf16 v16bf;
typedef __attribute__((ext_vector_type(8)))  __bf16 v8bf;
typedef __attribute__((ext_vector_type(8)))  float  v8f;

// ---------------------------------------------------------------------------
// fp32 -> bf16 (RNE) conversion pass, two elements per thread, packed store.
// RELU folds the inter-layer activation into the conversion.
// ---------------------------------------------------------------------------
template<bool RELU>
__global__ void to_bf16(const float* __restrict__ in,
                        unsigned int* __restrict__ out, long long npairs) {
    long long i = (long long)blockIdx.x * blockDim.x + threadIdx.x;
    if (i >= npairs) return;
    float a = in[2 * i];
    float b = in[2 * i + 1];
    if (RELU) { a = fmaxf(a, 0.0f); b = fmaxf(b, 0.0f); }
    unsigned ua = __float_as_uint(a), ub = __float_as_uint(b);
    unsigned ra = (ua + 0x7FFFu + ((ua >> 16) & 1u)) >> 16;   // round-NE
    unsigned rb = (ub + 0x7FFFu + ((ub >> 16) & 1u)) >> 16;
    out[i] = (ra & 0xFFFFu) | (rb << 16);
}

// Load one lane's slice of a 16x32 bf16 WMMA operand from a row-major bf16
// matrix. p points at row*K + kbase + hi*8: halves 0..7 from p[0..7],
// halves 8..15 from p[16..23] (ISA 7.12.2). Two aligned 16B loads.
static __device__ __forceinline__ v16bf load_k32_bf(const unsigned short* __restrict__ p) {
    v8bf lo = *(const v8bf*)p;
    v8bf hi = *(const v8bf*)(p + 16);
    v16bf v;
#pragma unroll
    for (int i = 0; i < 8; ++i) { v[i] = lo[i]; v[i + 8] = hi[i]; }
    return v;
}

// H[M,N] = Xb[M,K] @ Wb[N,K]^T, bf16 in / f32 out.
// One wave computes a full 16 x N strip: A slice is loaded once per K-step
// and reused across all N/16 WMMAs (8x less A traffic, 32 wmma per wave for
// N=128). Block = 8 waves; whole-wave early-exit keeps EXEC all-ones for the
// waves that do execute WMMA. M%16==0, K%32==0.
template<int N, int K>
__global__ __launch_bounds__(256)
void gemm_wmma_bf16(const unsigned short* __restrict__ Xb,
                    const unsigned short* __restrict__ Wb,
                    float* __restrict__ H, int mtiles) {
    const int lane  = threadIdx.x & 31;
    const int wave  = threadIdx.x >> 5;
    const int mtile = blockIdx.x * 8 + wave;
    if (mtile >= mtiles) return;
    const int r  = lane & 15;
    const int hi = lane >> 4;

    const unsigned short* xrow = Xb + (size_t)(mtile * 16 + r) * K + hi * 8;
    const unsigned short* wrow = Wb + (size_t)r * K + hi * 8;

    constexpr int NT = N / 16;
    v8f acc[NT];
    v8f zero = {};
#pragma unroll
    for (int nt = 0; nt < NT; ++nt) acc[nt] = zero;

#pragma unroll
    for (int kb = 0; kb < K; kb += 32) {
        v16bf a = load_k32_bf(xrow + kb);
#pragma unroll
        for (int nt = 0; nt < NT; ++nt) {
            v16bf b = load_k32_bf(wrow + (size_t)nt * 16 * K + kb);
            acc[nt] = __builtin_amdgcn_wmma_f32_16x16x32_bf16(
                          false, a, false, b, (short)0, acc[nt], false, false);
        }
    }

    // D layout: VGPR v -> row (v + 8*hi), column (lane & 15) within tile.
    float* obase = H + (size_t)(mtile * 16 + hi * 8) * N + r;
#pragma unroll
    for (int nt = 0; nt < NT; ++nt)
#pragma unroll
        for (int v = 0; v < 8; ++v)
            obase[(size_t)v * N + nt * 16] = acc[nt][v];
}

// ---------------------------------------------------------------------------
// Graph-side kernels
// ---------------------------------------------------------------------------

__global__ void init_deg(float* __restrict__ deg, int n) {
    int i = blockIdx.x * blockDim.x + threadIdx.x;
    if (i < n) deg[i] = 1.0f;                       // self-loop
}

__global__ void accum_deg(const long long* __restrict__ dst,
                          float* __restrict__ deg, int E) {
    int e = blockIdx.x * blockDim.x + threadIdx.x;
    if (e < E) atomicAdd(&deg[dst[e]], 1.0f);
}

__global__ void make_dinv(const float* __restrict__ deg,
                          float* __restrict__ dinv, int n) {
    int i = blockIdx.x * blockDim.x + threadIdx.x;
    if (i < n) {
        float d = deg[i];
        dinv[i] = d > 0.0f ? rsqrtf(d) : 0.0f;
    }
}

// out[i,c..c+3] = pre[i,c..c+3] * dinv[i]^2 + bias[c..c+3]
template<int C>
__global__ void self_bias(const float* __restrict__ pre,
                          const float* __restrict__ dinv,
                          const float* __restrict__ bias,
                          float* __restrict__ out, int n) {
    constexpr int CQ = C / 4;
    size_t i = (size_t)blockIdx.x * blockDim.x + threadIdx.x;   // quad index
    if (i >= (size_t)n * CQ) return;
    int node = (int)(i / CQ);
    int cq   = (int)(i % CQ);
    float di = dinv[node];
    float s  = di * di;
    float4 p = ((const float4*)pre)[i];
    float4 b = ((const float4*)bias)[cq];
    float4 o = { p.x * s + b.x, p.y * s + b.y, p.z * s + b.z, p.w * s + b.w };
    ((float4*)out)[i] = o;
}

// One wave per edge; lane owns C/32 channels (vector gather, scalar f32
// atomics into out[dst]). The 51MB feature tables sit in the 192MB L2.
template<int PER>   // channels per lane; C = 32*PER (PER = 4 or 2)
__global__ void edge_scatter(const long long* __restrict__ src,
                             const long long* __restrict__ dst,
                             const float* __restrict__ dinv,
                             const float* __restrict__ pre,
                             float* __restrict__ out, int E) {
    constexpr int C = 32 * PER;
    int wid  = (int)(((size_t)blockIdx.x * blockDim.x + threadIdx.x) >> 5);
    int lane = threadIdx.x & 31;
    if (wid >= E) return;
    long long s = src[wid];
    long long d = dst[wid];
    float norm = dinv[s] * dinv[d];
    const float* hp = pre + (size_t)s * C + lane * PER;
    float*       op = out + (size_t)d * C + lane * PER;
    float vals[PER];
    if constexpr (PER == 4) {
        float4 t = *(const float4*)hp;
        vals[0] = t.x; vals[1] = t.y; vals[2] = t.z; vals[3] = t.w;
    } else {
        float2 t = *(const float2*)hp;
        vals[0] = t.x; vals[1] = t.y;
    }
#pragma unroll
    for (int j = 0; j < PER; ++j)
        atomicAdd(&op[j], vals[j] * norm);
}

// ---------------------------------------------------------------------------
// Launch
// ---------------------------------------------------------------------------

extern "C" void kernel_launch(void* const* d_in, const int* in_sizes, int n_in,
                              void* d_out, int out_size, void* d_ws, size_t ws_size,
                              hipStream_t stream) {
    const float*     x    = (const float*)d_in[0];
    const long long* eidx = (const long long*)d_in[1];
    const float*     W1   = (const float*)d_in[2];
    const float*     b1   = (const float*)d_in[3];
    const float*     W2   = (const float*)d_in[4];
    const float*     b2   = (const float*)d_in[5];
    const float*     W3   = (const float*)d_in[6];
    const float*     b3   = (const float*)d_in[7];

    const int n = in_sizes[0] / 128;      // 100000
    const int E = in_sizes[1] / 2;        // 1600000
    const long long* src = eidx;
    const long long* dst = eidx + E;

    // Workspace layout:
    //   deg[n] f32 | dinv[n] f32 | bufA[n*128] f32 | bufB[n*128] f32 |
    //   xb[n*128] bf16 | wb[128*128] bf16
    float* deg  = (float*)d_ws;
    float* dinv = deg  + n;
    float* bufA = dinv + n;
    float* bufB = bufA + (size_t)n * 128;
    unsigned int*   xb32 = (unsigned int*)(bufB + (size_t)n * 128);
    unsigned short* xb   = (unsigned short*)xb32;
    unsigned int*   wb32 = xb32 + (size_t)n * 64;     // n*128 halves
    unsigned short* wb   = (unsigned short*)wb32;
    float* out = (float*)d_out;

    const int T = 256;
    const int mtiles = n / 16;                         // 6250
    const int gemm_blocks = (mtiles + 7) / 8;

    // --- degree / normalization (graph identical across layers) ------------
    init_deg <<<(n + T - 1) / T, T, 0, stream>>>(deg, n);
    accum_deg<<<(E + T - 1) / T, T, 0, stream>>>(dst, deg, E);
    make_dinv<<<(n + T - 1) / T, T, 0, stream>>>(deg, dinv, n);

    const long long xpairs  = (long long)n * 64;       // n*128/2
    const int xp_blocks     = (int)((xpairs + T - 1) / T);
    const int w128_blocks   = (128 * 128 / 2 + T - 1) / T;
    const int w64_blocks    = (64 * 128 / 2 + T - 1) / T;
    const int ew_blocks     = (int)(((size_t)E * 32 + T - 1) / T);
    const int nc128         = (int)(((size_t)n * 32 + T - 1) / T);   // quads
    const int nc64          = (int)(((size_t)n * 16 + T - 1) / T);

    // --- layer 1: pre1 = x @ W1^T ; h1 = agg(pre1) + b1 --------------------
    to_bf16<false><<<xp_blocks, T, 0, stream>>>(x, xb32, xpairs);
    to_bf16<false><<<w128_blocks, T, 0, stream>>>(W1, wb32, 128 * 128 / 2);
    gemm_wmma_bf16<128, 128><<<gemm_blocks, 256, 0, stream>>>(xb, wb, bufA, mtiles);
    self_bias<128><<<nc128, T, 0, stream>>>(bufA, dinv, b1, bufB, n);
    edge_scatter<4><<<ew_blocks, T, 0, stream>>>(src, dst, dinv, bufA, bufB, E);

    // --- layer 2: pre2 = relu(h1) @ W2^T ; h2 = agg(pre2) + b2 -------------
    to_bf16<true><<<xp_blocks, T, 0, stream>>>(bufB, xb32, xpairs);
    to_bf16<false><<<w128_blocks, T, 0, stream>>>(W2, wb32, 128 * 128 / 2);
    gemm_wmma_bf16<128, 128><<<gemm_blocks, 256, 0, stream>>>(xb, wb, bufA, mtiles);
    self_bias<128><<<nc128, T, 0, stream>>>(bufA, dinv, b2, bufB, n);
    edge_scatter<4><<<ew_blocks, T, 0, stream>>>(src, dst, dinv, bufA, bufB, E);

    // --- layer 3: pre3 = relu(h2) @ W3^T ; out = agg(pre3) + b3 ------------
    to_bf16<true><<<xp_blocks, T, 0, stream>>>(bufB, xb32, xpairs);
    to_bf16<false><<<w64_blocks, T, 0, stream>>>(W3, wb32, 64 * 128 / 2);
    gemm_wmma_bf16<64, 128><<<gemm_blocks, 256, 0, stream>>>(xb, wb, bufA, mtiles);
    self_bias<64><<<nc64, T, 0, stream>>>(bufA, dinv, b3, out, n);
    edge_scatter<2><<<ew_blocks, T, 0, stream>>>(src, dst, dinv, bufA, out, E);
}
